// CosineRouter_20306605375574
// MI455X (gfx1250) — matrix operand order repaired and checked by reading, not compiled, not run on hardware
//
#include <hip/hip_runtime.h>
#include <hip/hip_bf16.h>

typedef __attribute__((ext_vector_type(16))) __bf16 bf16x16;
typedef __attribute__((ext_vector_type(8)))  float  f32x8;
typedef unsigned int v4u __attribute__((ext_vector_type(4)));
typedef int          v8i __attribute__((ext_vector_type(8)));
typedef int          v4i __attribute__((ext_vector_type(4)));

#define T_TOK   16384
#define D_DIM   2048
#define E_EXP   64
#define P_PROT  2
#define NPROT   (E_EXP * P_PROT)   // 128 GEMM columns
#define SCALE_F 10.0f
#define EPS_F   1e-6f

#define TOK_PER_BLK 128            // 8 waves x 16-token m-tiles
#define KCHUNK      32

__device__ __forceinline__ unsigned short f2bf(float f) {
    unsigned u = __float_as_uint(f);
    u += 0x7FFFu + ((u >> 16) & 1u);
    return (unsigned short)(u >> 16);
}

// ---------------- kernel 1: inv L2 norm of each token row of h ----------------
__global__ void k_inv_norm_h(const float* __restrict__ h, float* __restrict__ invH) {
    const int wave = threadIdx.x >> 5;
    const int lane = threadIdx.x & 31;
    const int tok  = blockIdx.x * 8 + wave;
    const float* row = h + (long)tok * D_DIM;
    float s = 0.f;
#pragma unroll 4
    for (int k = lane * 4; k < D_DIM; k += 128) {
        float4 v = *(const float4*)(row + k);
        s += v.x * v.x + v.y * v.y + v.z * v.z + v.w * v.w;
    }
#pragma unroll
    for (int off = 16; off >= 1; off >>= 1) s += __shfl_xor(s, off, 32);
    if (lane == 0) invH[tok] = 1.f / (sqrtf(s) + EPS_F);
}

// -------- kernel 2: normalize prototypes, pack bf16 row-major [128][2048] ----
__global__ void k_norm_pack_p(const float* __restrict__ p, unsigned short* __restrict__ pbf) {
    const int wave = threadIdx.x >> 5;
    const int lane = threadIdx.x & 31;
    const int r    = blockIdx.x * 8 + wave;       // proto row 0..127
    const float* row = p + (long)r * D_DIM;
    float s = 0.f;
#pragma unroll 4
    for (int k = lane * 4; k < D_DIM; k += 128) {
        float4 v = *(const float4*)(row + k);
        s += v.x * v.x + v.y * v.y + v.z * v.z + v.w * v.w;
    }
#pragma unroll
    for (int off = 16; off >= 1; off >>= 1) s += __shfl_xor(s, off, 32);
    const float inv = 1.f / (sqrtf(s) + EPS_F);
#pragma unroll 4
    for (int k = lane * 4; k < D_DIM; k += 128) {
        float4 v = *(const float4*)(row + k);
        uint2 o;
        o.x = (unsigned)f2bf(v.x * inv) | ((unsigned)f2bf(v.y * inv) << 16);
        o.y = (unsigned)f2bf(v.z * inv) | ((unsigned)f2bf(v.w * inv) << 16);
        *(uint2*)(pbf + (long)r * D_DIM + k) = o;
    }
}

// ----- TDM: DMA one 128-row x 32-col bf16 tile of B (row stride 2048) to LDS --
__device__ __forceinline__ void tdm_load_b_chunk(const unsigned short* gsrc,
                                                 unsigned lds_addr) {
    unsigned long long ga = (unsigned long long)(const void*)gsrc;
    v4u g0;
    g0[0] = 1u;                                        // count=1, user descriptor
    g0[1] = lds_addr;                                  // LDS byte address
    g0[2] = (unsigned)(ga & 0xFFFFFFFFu);              // global_addr[31:0]
    g0[3] = (unsigned)((ga >> 32) & 0x01FFFFFFu)       // global_addr[56:32]
          | (2u << 30);                                // type = 2 ("image")
    v8i g1;
    g1[0] = (int)(1u << 16);                           // data_size=1 (2 bytes)
    g1[1] = (int)(((unsigned)D_DIM & 0xFFFFu) << 16);  // tensor_dim0[15:0]
    g1[2] = (int)(((unsigned)D_DIM >> 16)              // tensor_dim0[31:16]
          | ((unsigned)NPROT << 16));                  // tensor_dim1[15:0]
    g1[3] = (int)(((unsigned)NPROT >> 16)              // tensor_dim1[31:16]
          | ((unsigned)KCHUNK << 16));                 // tile_dim0 = 32
    g1[4] = (int)(unsigned)NPROT;                      // tile_dim1 = 128, tile_dim2 = 0
    g1[5] = (int)D_DIM;                                // tensor_dim0_stride[31:0]
    g1[6] = 0;                                         // stride hi / dim1_stride lo
    g1[7] = 0;
    v4i z4 = {0, 0, 0, 0};
    v8i z8 = {0, 0, 0, 0, 0, 0, 0, 0};
    __builtin_amdgcn_tensor_load_to_lds(g0, g1, z4, z4, z8, 0);
}

// ---------------- kernel 3: fused GEMM (WMMA bf16) + router math -------------
__global__ void __launch_bounds__(256)
k_router(const float* __restrict__ h, const unsigned short* __restrict__ pbf,
         const float* __restrict__ invH, const float* __restrict__ u,
         const int* __restrict__ kptr, float* __restrict__ out) {
    __shared__ __align__(16) unsigned short sB[2][NPROT * KCHUNK]; // 2 x 8 KB
    __shared__ __align__(16) float sSims[TOK_PER_BLK * NPROT];     // 64 KB

    const int tid  = threadIdx.x;
    const int wave = tid >> 5;
    const int lane = tid & 31;
    const int tr   = lane & 15;      // token-row in m-tile / col in n-tile
    const int grp  = lane >> 4;      // lane half -> K sub-groups
    const int token = blockIdx.x * TOK_PER_BLK + wave * 16 + tr;
    const float inv = invH[token];
    const float* hrow = h + (long)token * D_DIM;

    const unsigned ldsB0 = (unsigned)(size_t)&sB[0][0];
    const unsigned ldsB1 = (unsigned)(size_t)&sB[1][0];

    f32x8 acc[8];
#pragma unroll
    for (int n = 0; n < 8; ++n)
#pragma unroll
        for (int j = 0; j < 8; ++j) acc[n][j] = 0.f;

    // prologue: DMA chunk 0 into buffer 0
    if (wave == 0) {
        tdm_load_b_chunk(pbf, ldsB0);
        __builtin_amdgcn_s_wait_tensorcnt(0);
    }
    __syncthreads();

    int cur = 0;
    for (int kc = 0; kc < D_DIM; kc += KCHUNK) {
        // kick DMA of next chunk into the free buffer (overlaps with WMMAs)
        if (wave == 0 && kc + KCHUNK < D_DIM)
            tdm_load_b_chunk(pbf + kc + KCHUNK, cur ? ldsB0 : ldsB1);

        // A fragment (ISA 16-bit A layout), native f32->bf16 converts
        union { bf16x16 v; __bf16 e[16]; } A;
        {
            const float4* hp0 = (const float4*)(hrow + kc + grp * 8);
            const float4* hp1 = (const float4*)(hrow + kc + 16 + grp * 8);
            float4 f0 = hp0[0], f1 = hp0[1], f2 = hp1[0], f3 = hp1[1];
            A.e[0]  = (__bf16)(f0.x * inv); A.e[1]  = (__bf16)(f0.y * inv);
            A.e[2]  = (__bf16)(f0.z * inv); A.e[3]  = (__bf16)(f0.w * inv);
            A.e[4]  = (__bf16)(f1.x * inv); A.e[5]  = (__bf16)(f1.y * inv);
            A.e[6]  = (__bf16)(f1.z * inv); A.e[7]  = (__bf16)(f1.w * inv);
            A.e[8]  = (__bf16)(f2.x * inv); A.e[9]  = (__bf16)(f2.y * inv);
            A.e[10] = (__bf16)(f2.z * inv); A.e[11] = (__bf16)(f2.w * inv);
            A.e[12] = (__bf16)(f3.x * inv); A.e[13] = (__bf16)(f3.y * inv);
            A.e[14] = (__bf16)(f3.z * inv); A.e[15] = (__bf16)(f3.w * inv);
        }
        if (kc + KCHUNK < D_DIM)
            __builtin_prefetch(hrow + kc + KCHUNK, 0, 3);

        // load all 8 B fragments first so WMMAs aren't serialized on dscnt==0
        const unsigned short* sbc = &sB[cur][0];
        union BU { bf16x16 v; uint4 q[2]; } Bf[8];
#pragma unroll
        for (int n = 0; n < 8; ++n) {
            const uint4* bq = (const uint4*)(&sbc[(n * 16 + tr) * KCHUNK + grp * 8]);
            Bf[n].q[0] = bq[0];
            Bf[n].q[1] = bq[2];
        }
#pragma unroll
        for (int n = 0; n < 8; ++n)
            acc[n] = __builtin_amdgcn_wmma_f32_16x16x32_bf16(
                false, A.v, false, Bf[n].v, (short)0, acc[n], false, false);

        if (wave == 0) __builtin_amdgcn_s_wait_tensorcnt(0);
        __syncthreads();          // next buffer ready / current buffer released
        cur ^= 1;
    }

    // Spill sims to LDS (C layout: VGPR j -> M=j / j+8, lane -> N)
#pragma unroll
    for (int n = 0; n < 8; ++n) {
        const int col = n * 16 + tr;
#pragma unroll
        for (int j = 0; j < 8; ++j)
            sSims[(wave * 16 + j + grp * 8) * NPROT + col] = acc[n][j];
    }
    __syncthreads();

    // ---------------- phase 2: per-token router math ----------------
    if (tid < TOK_PER_BLK) {
        const int tok = blockIdx.x * TOK_PER_BLK + tid;
        float* srow = &sSims[tid * NPROT];
        const long TE = (long)T_TOK * E_EXP;
        float* outMask = out;
        float* outProb = out + TE;
        float* outLC   = out + 2 * TE;
        float* outLS   = out + 3 * TE;
        const long obase = (long)tok * E_EXP;

        float mx = -INFINITY;
        for (int e = 0; e < E_EXP; ++e) {
            float s0 = srow[2 * e], s1 = srow[2 * e + 1];
            float m = fmaxf(s0, s1);
            float lc = SCALE_F * (m + logf(expf(s0 - m) + expf(s1 - m)));
            srow[e] = lc;
            mx = fmaxf(mx, lc);
        }
        float sum = 0.f;
        for (int e = 0; e < E_EXP; ++e) {
            float lc = srow[e];
            sum += expf(lc - mx);
            float uv = u[obase + e];
            uv = fminf(fmaxf(uv, 1e-6f), 1.0f - 1e-6f);
            float g  = -logf(-logf(uv));
            float ls = lc + g;
            srow[E_EXP + e] = ls;
            outLC[obase + e] = lc;
            outLS[obase + e] = ls;
        }
        const float rsum = 1.f / sum;
        for (int e = 0; e < E_EXP; ++e)
            outProb[obase + e] = expf(srow[e] - mx) * rsum;

        int ksel = *kptr;
        if (ksel < 1) ksel = 1; if (ksel > E_EXP) ksel = E_EXP;
        unsigned long long selbits = 0ull;
        for (int i = 0; i < ksel; ++i) {
            int best = 0; float bv = -INFINITY;
            for (int e = 0; e < E_EXP; ++e) {
                if ((selbits >> e) & 1ull) continue;
                float v = srow[E_EXP + e];
                if (v > bv) { bv = v; best = e; }
            }
            selbits |= (1ull << best);
        }
        for (int e = 0; e < E_EXP; ++e)
            outMask[obase + e] = ((selbits >> e) & 1ull) ? 1.f : 0.f;
    }
}

extern "C" void kernel_launch(void* const* d_in, const int* in_sizes, int n_in,
                              void* d_out, int out_size, void* d_ws, size_t ws_size,
                              hipStream_t stream) {
    const float* h  = (const float*)d_in[0];
    const float* p  = (const float*)d_in[1];
    const float* u  = (const float*)d_in[2];
    const int*   kp = (const int*)d_in[3];

    char* ws = (char*)d_ws;
    float* invH         = (float*)ws;                    // 64 KB
    unsigned short* pbf = (unsigned short*)(ws + 65536); // 512 KB bf16 protos

    k_inv_norm_h<<<T_TOK / 8, 256, 0, stream>>>(h, invH);
    k_norm_pack_p<<<NPROT / 8, 256, 0, stream>>>(p, pbf);
    k_router<<<T_TOK / TOK_PER_BLK, 256, 0, stream>>>(h, pbf, invH, u, kp,
                                                      (float*)d_out);
}
